// model_network_21509196219098
// MI455X (gfx1250) — matrix operand order repaired and checked by reading, not compiled, not run on hardware
//
#include <hip/hip_runtime.h>
#include <hip/hip_bf16.h>
#include <math.h>

typedef __attribute__((ext_vector_type(16))) __bf16 v16bf;
typedef __attribute__((ext_vector_type(8)))  float  v8f;

#define WMMA_BF16(A, B, C) \
  __builtin_amdgcn_wmma_f32_16x16x32_bf16(false, (A), false, (B), (short)0, (C), false, false)

__device__ __forceinline__ float fast_rcp(float x) { return __builtin_amdgcn_rcpf(x); }
__device__ __forceinline__ float sigf(float x) {
  return fast_rcp(1.0f + __expf(-x));                 // v_exp_f32 + v_rcp_f32
}
__device__ __forceinline__ float tanh_fast(float x) {
  return 1.0f - 2.0f * fast_rcp(__expf(2.0f * x) + 1.0f);
}

// ---------------------------------------------------------------------------
// Network constants
// ---------------------------------------------------------------------------
#define NPOP_CELLS 768           // 256 + 256 + 128 + 128
#define T_STEPS 48
#define F_DIM 64
#define H_DIM 64
#define G4 256
#define O_DIM 10
#define K_WIN 10
#define N_OUTER 37               // t = 10 .. 46

__constant__ int c_pop_base[4] = {0, 256, 512, 640};
__constant__ int c_out_off[4]  = {199680, 49920, 149760, 0};   // pop -> flat out offset
__constant__ int c_pop_B[4]    = {256, 256, 128, 128};

// fragment-slot mapping for the 16-bit WMMA A layout:
//   k (0..63) -> kt = k>>5 ; k32 = k&31 ; half = (k32>>3)&1 ; e = (k32&7) + 8*(k32>>4)
__device__ __forceinline__ void a_slot(int k, int& kt, int& half, int& e) {
  kt = k >> 5;
  int k32 = k & 31;
  half = (k32 >> 3) & 1;
  e = (k32 & 7) + 8 * (k32 >> 4);
}

union V16U { v16bf v; uint4 q[2]; };

// ---------------------------------------------------------------------------
// Init: zero h/c state, soc ping-pong buffers, and the whole output tensor
// ---------------------------------------------------------------------------
__global__ __launch_bounds__(256)
void init_kernel(float* __restrict__ h_state, float* __restrict__ c_state,
                 float* __restrict__ soc_buf, float* __restrict__ out, int out_n) {
  int i = blockIdx.x * 256 + threadIdx.x;
  if (i < NPOP_CELLS * H_DIM) { h_state[i] = 0.0f; c_state[i] = 0.0f; }
  if (i < 2 * NPOP_CELLS) soc_buf[i] = 0.0f;
  if (i < out_n) out[i] = 0.0f;
}

// ---------------------------------------------------------------------------
// Kernel A: one outer step of the 4 LSTMs (all 768 cells), K=10 inner steps.
// 48 blocks x 128 threads; block = one 16-row cell tile; wave w owns gate
// columns [64w, 64w+64).  WMMA 16x16x32 bf16, f32 accumulate.
// x window and h are kept in LDS as pre-converted bf16 A-fragments.
// ---------------------------------------------------------------------------
__global__ __launch_bounds__(128)
void lstm_step_kernel(const float* __restrict__ sp_p,   // ws spikes [512][48][64]
                      const float* __restrict__ sp_b,   // ws spikes [256][48][64]
                      const float* __restrict__ Wih,    // [4][256][64]
                      const float* __restrict__ Whh,    // [4][256][64]
                      const float* __restrict__ bias,   // [4][256]
                      const float* __restrict__ fcW,    // [4][10][64]
                      const float* __restrict__ fcb,    // [4][10]
                      const float* __restrict__ scale,  // [4][64]
                      const float* __restrict__ smin,   // [4][64]
                      float* __restrict__ h_state,      // [768][64]
                      float* __restrict__ c_state,      // [768][64]
                      float* __restrict__ soc_buf,      // [2][768]
                      float* __restrict__ out,
                      int s, int t) {
  // xw[(step*2+kt)*2+half][m][e] : scaled spike window, bf16, A-fragment order
  __shared__ __align__(16) __bf16 xw[K_WIN * 4][16][16];   // 20 KB
  __shared__ __align__(16) __bf16 hb[4][16][16];           //  2 KB (kt*2+half)
  __shared__ float g_lds[16][G4];                          // 16 KB
  __shared__ float h_f32[16][H_DIM];                       //  4 KB

  const int gid0 = blockIdx.x * 16;
  const int pop  = (gid0 < 256) ? 0 : (gid0 < 512) ? 1 : (gid0 < 640) ? 2 : 3;

  const int tid  = threadIdx.x;
  const int w    = tid >> 5;
  const int lane = tid & 31;
  const int half = lane >> 4;
  const int lm   = lane & 15;

  const float* Wp  = Wih  + pop * G4 * F_DIM;
  const float* Whp = Whh  + pop * G4 * H_DIM;
  const float* bp  = bias + pop * G4;
  const float* scp = scale + pop * F_DIM;
  const float* smp = smin  + pop * F_DIM;

  // --- resident bf16 B-fragments of Wih / Whh (B[k][n] = W[n][k]) ----------
  v16bf fWih[4][2], fWhh[4][2];
  float bval[4];
#pragma unroll
  for (int nt = 0; nt < 4; ++nt) {
    const int n = w * 64 + nt * 16 + lm;
    bval[nt] = bp[n];
#pragma unroll
    for (int kt = 0; kt < 2; ++kt) {
      const int kb = kt * 32 + half * 8;
#pragma unroll
      for (int e = 0; e < 16; ++e) {
        fWih[nt][kt][e] = (__bf16)Wp[n * F_DIM + kb + e + 8 * (e >> 3)];
        fWhh[nt][kt][e] = (__bf16)Whp[n * H_DIM + kb + e + 8 * (e >> 3)];
      }
    }
  }

  // --- cooperative preload: scaled x window -> bf16 fragments in LDS -------
  for (int idx = tid; idx < 16 * K_WIN * F_DIM; idx += 128) {
    const int m    = idx / (K_WIN * F_DIM);
    const int rem  = idx % (K_WIN * F_DIM);
    const int step = rem / F_DIM;
    const int k    = rem % F_DIM;
    const int cell = gid0 + m;
    const int ts   = t - K_WIN + step;
    const float xv = (pop < 2)
        ? sp_p[((size_t)cell * T_STEPS + ts) * F_DIM + k]
        : sp_b[((size_t)(cell - 512) * T_STEPS + ts) * F_DIM + k];
    int kt, hf, e; a_slot(k, kt, hf, e);
    xw[(step * 2 + kt) * 2 + hf][m][e] = (__bf16)(xv * scp[k] + smp[k]);
  }
  // --- initial h (f32 + bf16 fragments) and per-thread c ------------------
  const int j  = tid & 63;
  const int m0 = tid >> 6;          // 0 or 1
  float c_reg[8];
#pragma unroll
  for (int q = 0; q < 8; ++q) {
    const int m = m0 + 2 * q;
    c_reg[q] = c_state[(gid0 + m) * H_DIM + j];
  }
  for (int idx = tid; idx < 16 * H_DIM; idx += 128) {
    const int m = idx >> 6, k = idx & 63;
    const float hv = h_state[gid0 * H_DIM + idx];
    h_f32[m][k] = hv;
    int kt, hf, e; a_slot(k, kt, hf, e);
    hb[kt * 2 + hf][m][e] = (__bf16)hv;
  }
  __syncthreads();

  for (int step = 0; step < K_WIN; ++step) {
    // A fragments straight from LDS (2x ds_load_b128 each, no conversion)
    V16U ax0, ax1, ah0, ah1;
    {
      const __bf16* px0 = &xw[(step * 2 + 0) * 2 + half][lm][0];
      const __bf16* px1 = &xw[(step * 2 + 1) * 2 + half][lm][0];
      const __bf16* ph0 = &hb[0 * 2 + half][lm][0];
      const __bf16* ph1 = &hb[1 * 2 + half][lm][0];
      ax0.q[0] = *(const uint4*)(px0);     ax0.q[1] = *(const uint4*)(px0 + 8);
      ax1.q[0] = *(const uint4*)(px1);     ax1.q[1] = *(const uint4*)(px1 + 8);
      ah0.q[0] = *(const uint4*)(ph0);     ah0.q[1] = *(const uint4*)(ph0 + 8);
      ah1.q[0] = *(const uint4*)(ph1);     ah1.q[1] = *(const uint4*)(ph1 + 8);
    }
    v8f acc[4];
#pragma unroll
    for (int nt = 0; nt < 4; ++nt) {
      const float bv = bval[nt];
      v8f c = {bv, bv, bv, bv, bv, bv, bv, bv};
      c = WMMA_BF16(ax0.v, fWih[nt][0], c);
      c = WMMA_BF16(ax1.v, fWih[nt][1], c);
      c = WMMA_BF16(ah0.v, fWhh[nt][0], c);
      c = WMMA_BF16(ah1.v, fWhh[nt][1], c);
      acc[nt] = c;
    }
    // C/D layout: vgpr r -> row m = r + 8*half, col n = lane%16
#pragma unroll
    for (int nt = 0; nt < 4; ++nt)
#pragma unroll
      for (int r = 0; r < 8; ++r)
        g_lds[r + 8 * half][w * 64 + nt * 16 + lm] = acc[nt][r];
    __syncthreads();

    // elementwise LSTM cell update (gate order i,f,g,o) -- fast transcendentals
    int kt, hf, e; a_slot(j, kt, hf, e);
#pragma unroll
    for (int q = 0; q < 8; ++q) {
      const int m = m0 + 2 * q;
      const float ig = sigf(g_lds[m][j]);
      const float fg = sigf(g_lds[m][64 + j]);
      const float gg = tanh_fast(g_lds[m][128 + j]);
      const float og = sigf(g_lds[m][192 + j]);
      const float cc = fg * c_reg[q] + ig * gg;
      c_reg[q] = cc;
      const float hv = og * tanh_fast(cc);
      h_f32[m][j] = hv;
      hb[kt * 2 + hf][m][e] = (__bf16)hv;
    }
    __syncthreads();
  }

  // --- write back state ----------------------------------------------------
#pragma unroll
  for (int q = 0; q < 8; ++q) {
    const int m = m0 + 2 * q;
    c_state[(gid0 + m) * H_DIM + j] = c_reg[q];
  }
  for (int idx = tid; idx < 16 * H_DIM; idx += 128)
    h_state[gid0 * H_DIM + idx] = h_f32[idx >> 6][idx & 63];

  // --- readout p = h @ fcW.T + fcb ; soc = p[:,0] --------------------------
  if (tid < 16 * O_DIM) {
    const int m = tid / O_DIM;
    const int o = tid % O_DIM;
    const float* fw = fcW + (pop * O_DIM + o) * H_DIM;
    float a = fcb[pop * O_DIM + o];
#pragma unroll 8
    for (int k = 0; k < H_DIM; ++k) a += h_f32[m][k] * fw[k];
    const int cellp = gid0 + m - c_pop_base[pop];
    out[c_out_off[pop] + (2 + s) * c_pop_B[pop] * O_DIM + cellp * O_DIM + o] = a;
    if (o == 0) soc_buf[(s & 1) * NPOP_CELLS + gid0 + m] = a;
  }
}

// ---------------------------------------------------------------------------
// Kernel B: spike masks + connectivity injection into spikes[t+1].
// 768 blocks (one per target cell) x 256 threads (8 waves, wave per f-row).
// float4 loads: L2-bandwidth-bound GEMV.
// ---------------------------------------------------------------------------
__global__ __launch_bounds__(256)
void conn_step_kernel(float* __restrict__ sp_p, float* __restrict__ sp_b,
                      const float* __restrict__ conn_p,   // [2][256][64][768]
                      const float* __restrict__ conn_b,   // [2][128][64][768]
                      const float* __restrict__ ei,       // [4][4]
                      const float* __restrict__ soc_buf,  // [2][768]
                      int s, int t) {
  __shared__ __align__(16) float wm[NPOP_CELLS];
  const int bid = blockIdx.x;
  const int tgt = (bid < 256) ? 0 : (bid < 512) ? 1 : (bid < 640) ? 2 : 3;
  const int cellt = bid - c_pop_base[tgt];

  const float* cur  = soc_buf + (s & 1) * NPOP_CELLS;
  const float* prev = soc_buf + ((s & 1) ^ 1) * NPOP_CELLS;
  for (int n = threadIdx.x; n < NPOP_CELLS; n += 256) {
    const int sp = (n < 256) ? 0 : (n < 512) ? 1 : (n < 640) ? 2 : 3;
    const bool fire = (cur[n] > 0.1f) && !(prev[n] > 0.1f);
    wm[n] = fire ? exp10f(ei[sp * 4 + tgt]) : 0.0f;
  }
  __syncthreads();

  const float* crow = (tgt < 2)
      ? conn_p + (size_t)(tgt * 256 + cellt) * F_DIM * NPOP_CELLS
      : conn_b + (size_t)((tgt - 2) * 128 + cellt) * F_DIM * NPOP_CELLS;
  float* dst = (tgt < 2) ? sp_p + ((size_t)bid * T_STEPS + (t + 1)) * F_DIM
                         : sp_b + ((size_t)(bid - 512) * T_STEPS + (t + 1)) * F_DIM;

  const int w = threadIdx.x >> 5;
  const int lane = threadIdx.x & 31;
  const float4* wm4 = (const float4*)wm;
  for (int f = w; f < F_DIM; f += 8) {
    const float4* row4 = (const float4*)(crow + (size_t)f * NPOP_CELLS);
    float a = 0.0f;
#pragma unroll
    for (int it = 0; it < NPOP_CELLS / (32 * 4); ++it) {
      const float4 r = row4[lane + 32 * it];
      const float4 m = wm4[lane + 32 * it];
      a += r.x * m.x + r.y * m.y + r.z * m.z + r.w * m.w;
    }
#pragma unroll
    for (int off = 16; off > 0; off >>= 1) a += __shfl_down(a, off, 32);
    if (lane == 0) dst[f] += a;
  }
}

// ---------------------------------------------------------------------------
// Host driver
// ---------------------------------------------------------------------------
extern "C" void kernel_launch(void* const* d_in, const int* in_sizes, int n_in,
                              void* d_out, int out_size, void* d_ws, size_t ws_size,
                              hipStream_t stream) {
  (void)in_sizes; (void)n_in; (void)ws_size;
  const float* in_sp_p  = (const float*)d_in[0];   // [2,256,48,64]
  const float* in_sp_b  = (const float*)d_in[1];   // [2,128,48,64]
  const float* Wih      = (const float*)d_in[2];
  const float* Whh      = (const float*)d_in[3];
  const float* bvec     = (const float*)d_in[4];
  const float* fcW      = (const float*)d_in[5];
  const float* fcb      = (const float*)d_in[6];
  const float* scl      = (const float*)d_in[7];
  const float* smn      = (const float*)d_in[8];
  const float* conn_p   = (const float*)d_in[9];
  const float* conn_b   = (const float*)d_in[10];
  const float* ei       = (const float*)d_in[11];
  float* out = (float*)d_out;

  // workspace layout (floats)
  float* ws   = (float*)d_ws;
  float* sp_p = ws;                       // 512*48*64   = 1,572,864
  float* sp_b = ws + 1572864;             // 256*48*64   =   786,432
  float* hst  = ws + 2359296;             // 768*64      =    49,152
  float* cst  = ws + 2408448;             // 768*64      =    49,152
  float* soc  = ws + 2457600;             // 2*768       =     1,536

  hipMemcpyAsync(sp_p, in_sp_p, (size_t)1572864 * sizeof(float),
                 hipMemcpyDeviceToDevice, stream);
  hipMemcpyAsync(sp_b, in_sp_b, (size_t)786432 * sizeof(float),
                 hipMemcpyDeviceToDevice, stream);

  const int out_n = out_size;                       // expect 299520
  const int init_blocks = (out_n + 255) / 256 > 192 ? (out_n + 255) / 256 : 192;
  init_kernel<<<init_blocks, 256, 0, stream>>>(hst, cst, soc, out, out_n);

  for (int s = 0; s < N_OUTER; ++s) {
    const int t = K_WIN + s;
    lstm_step_kernel<<<NPOP_CELLS / 16, 128, 0, stream>>>(
        sp_p, sp_b, Wih, Whh, bvec, fcW, fcb, scl, smn,
        hst, cst, soc, out, s, t);
    conn_step_kernel<<<NPOP_CELLS, 256, 0, stream>>>(
        sp_p, sp_b, conn_p, conn_b, ei, soc, s, t);
  }
}
